// AdditiveAttention_2937757630857
// MI455X (gfx1250) — matrix operand order repaired and checked by reading, not compiled, not run on hardware
//
#include <hip/hip_runtime.h>
#include <hip/hip_bf16.h>
#include <math.h>

// Problem constants (from reference): B=64, S=2048, QD=KD=HD=VD=1024
#define BB   64
#define SS   2048
#define DD   1024

typedef __attribute__((ext_vector_type(16))) _Float16 v16h;
typedef __attribute__((ext_vector_type(8)))  _Float16 v8h;
typedef __attribute__((ext_vector_type(8)))  float    v8f;
typedef __attribute__((ext_vector_type(4)))  float    v4f;

union HFrag {
    v16h v;
    v8h  h8[2];
};

// ---------------------------------------------------------------------------
// Fast tanh: use gfx1250 hardware v_tanh_f32 when clang exposes it; otherwise
// a branch-free 1 - 2/(exp(2x)+1) (single v_exp_f32, correct saturation at
// +/-inf, no EXEC divergence — unlike libm tanhf's branchy expansion).
// ---------------------------------------------------------------------------
__device__ __forceinline__ float fast_tanh(float x) {
#if __has_builtin(__builtin_amdgcn_tanhf)
    return __builtin_amdgcn_tanhf(x);
#elif __has_builtin(__builtin_amdgcn_tanh_f32)
    return __builtin_amdgcn_tanh_f32(x);
#else
    float t = __expf(2.0f * x);          // 0 .. +inf
    return 1.0f - 2.0f / (t + 1.0f);     // -1 .. +1, branch-free
#endif
}

// ---------------------------------------------------------------------------
// Kernel 0: zero the context region of d_out (atomicAdd target)
// ---------------------------------------------------------------------------
__global__ void zero_ctx_kernel(float* __restrict__ ctx, int n) {
    int i = blockIdx.x * blockDim.x + threadIdx.x;
    if (i < n) ctx[i] = 0.0f;
}

// ---------------------------------------------------------------------------
// Kernel 1: q_proj[b,h] = sum_k query[b,k]*Wq[k,h] + bq[h]   (64x1024 out)
// grid = 64 (one block per b), block = 256; Wq is L2-resident (4 MB).
// ---------------------------------------------------------------------------
__global__ void qproj_kernel(const float* __restrict__ query,
                             const float* __restrict__ Wq,
                             const float* __restrict__ bq,
                             float* __restrict__ qproj) {
    const int b   = blockIdx.x;
    const int tid = threadIdx.x;
    const float* q = query + (size_t)b * DD;
    for (int h = tid; h < DD; h += blockDim.x) {
        float acc = bq[h];
        #pragma unroll 4
        for (int k = 0; k < DD; ++k)
            acc = fmaf(q[k], Wq[(size_t)k * DD + h], acc);  // coalesced across threads
        qproj[(size_t)b * DD + h] = acc;
    }
}

// ---------------------------------------------------------------------------
// Kernel 2: WkT[h][k] = (f16) Wk[k][h]   (transpose + fp32->fp16)
// So each B-fragment lane load is two runs of 8 contiguous halves.
// ---------------------------------------------------------------------------
__global__ void wkT_kernel(const float* __restrict__ Wk,
                           _Float16* __restrict__ wkT) {
    int idx = blockIdx.x * blockDim.x + threadIdx.x;   // 1M elements
    if (idx >= DD * DD) return;
    int h = idx & (DD - 1);
    int k = idx >> 10;
    wkT[(size_t)h * DD + k] = (_Float16)Wk[(size_t)k * DD + h];  // coalesced read
}

// ---------------------------------------------------------------------------
// Kernel 3 (WMMA core):
// scores[b, s0+m] = sum_h tanh(qproj[b,h] + bk[h] + (keys[b,s0+m,:] @ Wk)[h]) * We[h] + be
//
// Block = 128 threads (4 waves, wave32).  Each block: one b, 16 s-rows.
// Keys tile (16x1024 f32, streamed non-temporal) -> f16 in LDS (row stride
// 1032 halves: conflict-free b128 fragment gathers).  Wave w handles
// h-tiles [16w, 16w+16).  Inner loop: 32 x v_wmma_f32_16x16x32_f16 (K=1024).
// ---------------------------------------------------------------------------
#define KROW 1032   // padded LDS row stride in halves (1024 + 8)

__global__ __launch_bounds__(128)
void scores_kernel(const float* __restrict__ keys,
                   const float* __restrict__ qproj,
                   const _Float16* __restrict__ wkT,
                   const float* __restrict__ bk,
                   const float* __restrict__ We,
                   const float* __restrict__ be,
                   float* __restrict__ scores) {
    __shared__ _Float16 kbuf[16 * KROW];     // ~33 KB
    __shared__ float    sred[4][16];

    const int b    = blockIdx.y;
    const int s0   = blockIdx.x * 16;
    const int tid  = threadIdx.x;
    const int lane = tid & 31;
    const int wid  = tid >> 5;

    // --- cooperative load: 16 rows x 1024 f32 -> f16 LDS tile (NT stream) ---
    const float* krow = keys + ((size_t)b * SS + s0) * DD;
    for (int i = tid; i < 16 * 256; i += 128) {          // 256 float4 per row
        int row = i >> 8;
        int c4  = (i & 255) << 2;
        v4f f = __builtin_nontemporal_load((const v4f*)(krow + (size_t)row * DD + c4));
        _Float16* d = &kbuf[row * KROW + c4];
        d[0] = (_Float16)f.x; d[1] = (_Float16)f.y;
        d[2] = (_Float16)f.z; d[3] = (_Float16)f.w;
    }
    __syncthreads();

    // fragment-layout helpers (CDNA5 16-bit WMMA layouts, wave32)
    const int n    = lane & 15;                  // C-tile column / B column / A row
    const int aoff = (lane < 16) ? 0 : 8;        // A: K 0-7/16-23 vs 8-15/24-31
    const int boff = (lane < 16) ? 0 : 8;        // B: K 0-7/16-23 vs 8-15/24-31

    float sacc[8];
    #pragma unroll
    for (int j = 0; j < 8; ++j) sacc[j] = 0.0f;

    const int ht0 = wid * 16;
    for (int ht = ht0; ht < ht0 + 16; ++ht) {
        const int h   = ht * 16 + n;
        const float qb = qproj[(size_t)b * DD + h] + bk[h];
        const float wE = We[h];
        const _Float16* wp = wkT + (size_t)h * DD;   // L2-resident (2 MB total)

        v8f c = {};
        #pragma unroll 4
        for (int kc = 0; kc < DD; kc += 32) {
            HFrag a, bm;
            const _Float16* ap = &kbuf[n * KROW + kc + aoff];
            a.h8[0]  = *(const v8h*)(ap);            // K kc+aoff    .. +7
            a.h8[1]  = *(const v8h*)(ap + 16);       // K kc+16+aoff .. +7
            bm.h8[0] = *(const v8h*)(wp + kc + boff);      // K kc+boff   .. +7
            bm.h8[1] = *(const v8h*)(wp + kc + boff + 16); // K kc+16+boff .. +7
            c = __builtin_amdgcn_wmma_f32_16x16x32_f16(
                    /*neg_a=*/false, a.v, /*neg_b=*/false, bm.v,
                    /*c_mod=*/(short)0, c, /*reuse_a=*/false, /*reuse_b=*/false);
        }
        // C layout: slot j, lanes 0-15 -> (M=j, N=lane); lanes 16-31 -> (M=j+8)
        #pragma unroll
        for (int j = 0; j < 8; ++j)
            sacc[j] = fmaf(fast_tanh(c[j] + qb), wE, sacc[j]);
    }

    // reduce over the 16 columns (lanes within each half-wave)
    #pragma unroll
    for (int j = 0; j < 8; ++j) {
        sacc[j] += __shfl_xor(sacc[j], 1, 16);
        sacc[j] += __shfl_xor(sacc[j], 2, 16);
        sacc[j] += __shfl_xor(sacc[j], 4, 16);
        sacc[j] += __shfl_xor(sacc[j], 8, 16);
    }
    if (n == 0) {                      // lanes 0 (rows 0-7) and 16 (rows 8-15)
        const int rbase = (lane < 16) ? 0 : 8;
        #pragma unroll
        for (int j = 0; j < 8; ++j) sred[wid][rbase + j] = sacc[j];
    }
    __syncthreads();
    if (tid < 16) {
        float s = sred[0][tid] + sred[1][tid] + sred[2][tid] + sred[3][tid] + be[0];
        scores[(size_t)b * SS + s0 + tid] = s;
    }
}

// ---------------------------------------------------------------------------
// Kernel 4: row softmax over S=2048  (one block per batch row)
// ---------------------------------------------------------------------------
__global__ void softmax_kernel(const float* __restrict__ scores,
                               float* __restrict__ weights) {
    __shared__ float red[256];
    const int b   = blockIdx.x;
    const int tid = threadIdx.x;
    const float* sc = scores + (size_t)b * SS;
    float* w = weights + (size_t)b * SS;

    float mx = -3.402823466e38f;
    for (int s = tid; s < SS; s += 256) mx = fmaxf(mx, sc[s]);
    red[tid] = mx; __syncthreads();
    for (int off = 128; off; off >>= 1) {
        if (tid < off) red[tid] = fmaxf(red[tid], red[tid + off]);
        __syncthreads();
    }
    mx = red[0];
    __syncthreads();

    float sum = 0.0f;
    for (int s = tid; s < SS; s += 256) {
        float e = __expf(sc[s] - mx);
        w[s] = e;
        sum += e;
    }
    red[tid] = sum; __syncthreads();
    for (int off = 128; off; off >>= 1) {
        if (tid < off) red[tid] += red[tid + off];
        __syncthreads();
    }
    const float inv = 1.0f / red[0];
    for (int s = tid; s < SS; s += 256) w[s] *= inv;
}

// ---------------------------------------------------------------------------
// Kernel 5: context[b,v] = sum_s weights[b,s] * values[b,s,v]
// grid = (4 v-chunks, 8 s-chunks, 64 b), block = 256.  Streams 512 MB of
// values fully coalesced + non-temporal; partials combined with f32 atomics.
// ---------------------------------------------------------------------------
__global__ void context_kernel(const float* __restrict__ weights,
                               const float* __restrict__ values,
                               float* __restrict__ ctx) {
    const int v  = blockIdx.x * 256 + threadIdx.x;
    const int s0 = blockIdx.y * 256;
    const int b  = blockIdx.z;

    const float* vb = values + ((size_t)b * SS + s0) * DD + v;
    const float* wb = weights + (size_t)b * SS + s0;

    float acc = 0.0f;
    #pragma unroll 4
    for (int i = 0; i < 256; ++i)
        acc = fmaf(wb[i], __builtin_nontemporal_load(vb + (size_t)i * DD), acc);
    atomicAdd(&ctx[(size_t)b * DD + v], acc);
}

// ---------------------------------------------------------------------------
// Host launcher
// Inputs: 0 query[64,1024] 1 keys[64,2048,1024] 2 values[64,2048,1024]
//         3 Wq[1024,1024] 4 bq[1024] 5 Wk[1024,1024] 6 bk[1024]
//         7 We[1024] 8 be[1]
// Output: context[64,1024] (65536 f32)  ++  weights[64,2048] (131072 f32)
// ---------------------------------------------------------------------------
extern "C" void kernel_launch(void* const* d_in, const int* in_sizes, int n_in,
                              void* d_out, int out_size, void* d_ws, size_t ws_size,
                              hipStream_t stream) {
    const float* query  = (const float*)d_in[0];
    const float* keys   = (const float*)d_in[1];
    const float* values = (const float*)d_in[2];
    const float* Wq     = (const float*)d_in[3];
    const float* bq     = (const float*)d_in[4];
    const float* Wk     = (const float*)d_in[5];
    const float* bk     = (const float*)d_in[6];
    const float* We     = (const float*)d_in[7];
    const float* be     = (const float*)d_in[8];

    float* out_ctx = (float*)d_out;                       // [64,1024]
    float* out_w   = out_ctx + (size_t)BB * DD;           // [64,2048]

    // workspace layout (bytes): qproj 256KB | WkT f16 2MB | scores 512KB
    char* ws = (char*)d_ws;
    float*    qproj  = (float*)ws;                                     // 262144 B
    _Float16* wkT    = (_Float16*)(ws + 262144);                       // 2097152 B
    float*    scores = (float*)(ws + 262144 + 2097152);                // 524288 B

    zero_ctx_kernel<<<(BB * DD + 255) / 256, 256, 0, stream>>>(out_ctx, BB * DD);

    qproj_kernel<<<BB, 256, 0, stream>>>(query, Wq, bq, qproj);

    wkT_kernel<<<(DD * DD + 255) / 256, 256, 0, stream>>>(Wk, wkT);

    scores_kernel<<<dim3(SS / 16, BB), 128, 0, stream>>>(
        keys, qproj, wkT, bk, We, be, scores);

    softmax_kernel<<<BB, 256, 0, stream>>>(scores, out_w);

    context_kernel<<<dim3(DD / 256, SS / 256, BB), 256, 0, stream>>>(
        out_w, values, out_ctx);
}
